// CapsuleNetwork_5858335392090
// MI455X (gfx1250) — matrix operand order repaired and checked by reading, not compiled, not run on hardware
//
#include <hip/hip_runtime.h>
#include <hip/hip_bf16.h>
#include <math.h>

// ---------------------------------------------------------------------------
// CapsNet forward for MI455X (gfx1250, wave32, WMMA)
//   conv1 (direct VALU, 0.26 GF) -> prim conv as implicit GEMM
//   (M=576,N=8192,K=20736; v_wmma_f32_16x16x32_f16, 195.7 GF, 679MB weights)
//   -> squash -> 3x dynamic routing with u_hat recomputed on the fly via
//   v_wmma_f32_16x16x4_f32 (coef folded into A fragment).
// ---------------------------------------------------------------------------

typedef __attribute__((ext_vector_type(16))) _Float16 v16h;
typedef __attribute__((ext_vector_type(8)))  float    v8f;
typedef __attribute__((ext_vector_type(2)))  float    v2f;

union HFrag { v16h v; unsigned u[8]; };

__device__ __forceinline__ unsigned pack_f16x2(float a, float b) {
  union { _Float16 h[2]; unsigned u; } p;
  p.h[0] = (_Float16)a; p.h[1] = (_Float16)b;
  return p.u;
}

#if defined(__has_builtin)
#if __has_builtin(__builtin_amdgcn_wmma_f32_16x16x4_f32)
#define HAVE_WMMA_F32X4 1
#endif
#endif

// D = A(16x4 f32) x B(4x16 f32) + C, wave32 WMMA layouts:
//  A: lanes0-15 row=lane, V0=K0,V1=K1 ; lanes16-31 V0=K2,V1=K3
//  B: lanes0-15 col=lane, V0=K0,V1=K1 ; lanes16-31 V0=K2,V1=K3
__device__ __forceinline__ v8f wmma_f32_k4(v2f a, v2f b, v8f c) {
#if defined(HAVE_WMMA_F32X4)
  return __builtin_amdgcn_wmma_f32_16x16x4_f32(false, a, false, b,
                                               (short)0, c, false, false);
#else
  const int lane = __lane_id();
  const int n = lane & 15;
  const int half = lane >> 4;
  const float b0 = __shfl(b.x, n, 32);
  const float b1 = __shfl(b.y, n, 32);
  const float b2 = __shfl(b.x, n + 16, 32);
  const float b3 = __shfl(b.y, n + 16, 32);
#pragma unroll
  for (int j = 0; j < 8; ++j) {
    const int m = j + 8 * half;
    const float a0 = __shfl(a.x, m, 32);
    const float a1 = __shfl(a.y, m, 32);
    const float a2 = __shfl(a.x, m + 16, 32);
    const float a3 = __shfl(a.y, m + 16, 32);
    c[j] += a0 * b0 + a1 * b1 + a2 * b2 + a3 * b3;
  }
  return c;
#endif
}

// ---------------------------------------------------------------------------
// conv1: x[16,1,28,28] * w[256,1,9,9] + b, ReLU -> h[16,256,20,20]
// ---------------------------------------------------------------------------
__global__ __launch_bounds__(256)
void conv1_kernel(const float* __restrict__ X, const float* __restrict__ W,
                  const float* __restrict__ Bv, float* __restrict__ H) {
  const int idx = blockIdx.x * 256 + threadIdx.x;     // 16*256*400
  if (idx >= 1638400) return;
  const int x  = idx % 20;
  const int y  = (idx / 20) % 20;
  const int oc = (idx / 400) % 256;
  const int b  = idx / 102400;
  const float* xp = X + b * 784;
  const float* wp = W + oc * 81;
  float s = Bv[oc];
#pragma unroll
  for (int ky = 0; ky < 9; ++ky)
#pragma unroll
    for (int kx = 0; kx < 9; ++kx)
      s += xp[(y + ky) * 28 + (x + kx)] * wp[ky * 9 + kx];
  H[idx] = fmaxf(s, 0.f);
}

// ---------------------------------------------------------------------------
// prim conv as implicit GEMM: out[b][n][pix] = sum_k A[m][k] * Wt[n][k] + bias
//   m = b*36+pix (M=576), k = c*81+ky*9+kx (K=20736), n filter (N=8192)
// Block: 256 thr (8 waves). Tile M=64 (LDS, shared), each wave owns 16 N cols.
// ---------------------------------------------------------------------------
__global__ __launch_bounds__(256)
void prim_gemm_kernel(const float* __restrict__ H, const float* __restrict__ W,
                      const float* __restrict__ Bv, float* __restrict__ OUT) {
  __shared__ _Float16 Ah[64 * 34];                    // pad 34 -> conflict-free
  const int lane = threadIdx.x & 31;
  const int wid  = threadIdx.x >> 5;
  const int half = lane >> 4;
  const int l16  = lane & 15;
  const int n    = blockIdx.x * 128 + wid * 16 + l16; // this lane's column
  const int mBase = blockIdx.y * 64;

  // cooperative A-load coordinates: thread -> (row 0..63, 8 consecutive k)
  const int row = threadIdx.x >> 2;
  const int kq  = (threadIdx.x & 3) * 8;
  const int mg  = mBase + row;
  const int bb  = mg / 36;
  const int pix = mg % 36;
  const int oy = pix / 6, ox = pix % 6;

  const float* wp = W + (size_t)n * 20736;

  v8f acc[4] = {};
  for (int kb = 0; kb < 648; ++kb) {
    const int K0 = kb * 32;
    {   // stage A tile (64x32 f32 -> f16) into LDS
      int k0 = K0 + kq;
      int c  = k0 / 81;
      int rem = k0 - c * 81;
      int ky = rem / 9;
      int kx = rem - ky * 9;
#pragma unroll
      for (int e = 0; e < 8; ++e) {
        const int y = oy * 2 + ky, x = ox * 2 + kx;
        Ah[row * 34 + kq + e] = (_Float16)H[((bb * 256 + c) * 20 + y) * 20 + x];
        if (++kx == 9) { kx = 0; if (++ky == 9) { ky = 0; ++c; } }
      }
    }
    __syncthreads();

    // B fragment (32x16 f16): lane col=l16, K = half*16 + {2j,2j+1}
    HFrag bf;
#pragma unroll
    for (int j = 0; j < 8; ++j) {
      const int kl = half * 16 + 2 * j;
      const float2 w2 = *(const float2*)(wp + K0 + kl);
      bf.u[j] = pack_f16x2(w2.x, w2.y);
    }
    __builtin_prefetch(wp + K0 + 512, 0, 1);          // global_prefetch_b8

#pragma unroll
    for (int f = 0; f < 4; ++f) {
      // A fragment (16x32 f16): lane row=l16, K = half*16 + {2j,2j+1}
      HFrag af;
#pragma unroll
      for (int j = 0; j < 8; ++j) {
        const int kl = half * 16 + 2 * j;
        af.u[j] = *(const unsigned*)&Ah[(f * 16 + l16) * 34 + kl];
      }
      acc[f] = __builtin_amdgcn_wmma_f32_16x16x32_f16(
          false, af.v, false, bf.v, (short)0, acc[f], false, false);
    }
    __syncthreads();
  }

  const float bv = Bv[n];
#pragma unroll
  for (int f = 0; f < 4; ++f)
#pragma unroll
    for (int j = 0; j < 8; ++j) {
      const int mloc = mBase + f * 16 + j + 8 * half;  // D row
      const int ob = mloc / 36, op = mloc % 36;
      OUT[((size_t)(ob * 8192 + n)) * 36 + op] = acc[f][j] + bv;
    }
}

// ---------------------------------------------------------------------------
// squash u in place: 589824 capsules of 8 floats
// ---------------------------------------------------------------------------
__global__ __launch_bounds__(256)
void squash_kernel(float* __restrict__ U) {
  const int g = blockIdx.x * 256 + threadIdx.x;
  if (g >= 589824) return;
  float* p = U + (size_t)g * 8;
  float v[8], sq = 0.f;
#pragma unroll
  for (int i = 0; i < 8; ++i) { v[i] = p[i]; sq += v[i] * v[i]; }
  const float sc = (sq / (1.f + sq)) / sqrtf(sq + 1e-8f);
#pragma unroll
  for (int i = 0; i < 8; ++i) p[i] = v[i] * sc;
}

__global__ __launch_bounds__(256)
void zero_kernel(float* __restrict__ P, int n) {
  const int g = blockIdx.x * 256 + threadIdx.x;
  if (g < n) P[g] = 0.f;
}

// ---------------------------------------------------------------------------
// softmax stats over R per (b,c): max and 1/sum(exp)
// ---------------------------------------------------------------------------
__global__ __launch_bounds__(256)
void softmax_stats_kernel(const float* __restrict__ BL, float* __restrict__ ST) {
  __shared__ float red[256];
  const float* p = BL + (size_t)blockIdx.x * 36864;
  float m = -3.4e38f;
  for (int i = threadIdx.x; i < 36864; i += 256) m = fmaxf(m, p[i]);
  red[threadIdx.x] = m; __syncthreads();
  for (int s = 128; s > 0; s >>= 1) {
    if (threadIdx.x < s) red[threadIdx.x] = fmaxf(red[threadIdx.x], red[threadIdx.x + s]);
    __syncthreads();
  }
  const float mx = red[0];
  __syncthreads();
  float sum = 0.f;
  for (int i = threadIdx.x; i < 36864; i += 256) sum += __expf(p[i] - mx);
  red[threadIdx.x] = sum; __syncthreads();
  for (int s = 128; s > 0; s >>= 1) {
    if (threadIdx.x < s) red[threadIdx.x] += red[threadIdx.x + s];
    __syncthreads();
  }
  if (threadIdx.x == 0) { ST[blockIdx.x * 2] = mx; ST[blockIdx.x * 2 + 1] = 1.f / red[0]; }
}

// ---------------------------------------------------------------------------
// s-pass: s[b,c,o] = sum_r softmax(BL)[b,c,r] * (u[b,r,:] @ W[r,c,:,:])
// u_hat recomputed via f32 WMMA (K=4 x2); coef folded into A rows.
// 256 blocks x 8 waves, 18 r per wave; block partials via ds_add_f32.
// ---------------------------------------------------------------------------
__global__ __launch_bounds__(256)
void route_s_kernel(const float* __restrict__ U, const float* __restrict__ RW,
                    const float* __restrict__ BL, const float* __restrict__ ST,
                    float* __restrict__ PART) {
  __shared__ float s_lds[2560];
  for (int i = threadIdx.x; i < 2560; i += 256) s_lds[i] = 0.f;
  __syncthreads();

  const int lane = threadIdx.x & 31;
  const int wid  = threadIdx.x >> 5;
  const int half = lane >> 4;
  const int bb   = lane & 15;          // A row (batch) & B/D column index

  float mx[10], inv[10];
#pragma unroll
  for (int c = 0; c < 10; ++c) {
    mx[c]  = ST[(bb * 10 + c) * 2];
    inv[c] = ST[(bb * 10 + c) * 2 + 1];
  }

  v8f acc[10] = {};
  const int gw = blockIdx.x * 8 + wid;
  for (int t = 0; t < 18; ++t) {
    const int r = gw * 18 + t;
    const float* ub = U + (size_t)bb * 294912 + (size_t)r * 8;
    const float2 a1 = *(const float2*)(ub + 2 * half);      // K 0..3 slice
    const float2 a2 = *(const float2*)(ub + 4 + 2 * half);  // K 4..7 slice
    const float* wb = RW + (size_t)r * 1280;                // [c][i][o]
#pragma unroll
    for (int c = 0; c < 10; ++c) {
      const float blv  = BL[((size_t)(bb * 10 + c)) * 36864 + r];
      const float coef = __expf(blv - mx[c]) * inv[c];
      const float* wc = wb + c * 128;
      const int i0 = 2 * half;
      const v2f B1 = { wc[i0 * 16 + bb],       wc[(i0 + 1) * 16 + bb] };
      const v2f B2 = { wc[(i0 + 4) * 16 + bb], wc[(i0 + 5) * 16 + bb] };
      const v2f A1 = { a1.x * coef, a1.y * coef };
      const v2f A2 = { a2.x * coef, a2.y * coef };
      acc[c] = wmma_f32_k4(A1, B1, acc[c]);
      acc[c] = wmma_f32_k4(A2, B2, acc[c]);
    }
  }

#pragma unroll
  for (int c = 0; c < 10; ++c)
#pragma unroll
    for (int j = 0; j < 8; ++j) {
      const int brow = j + 8 * half;
      atomicAdd(&s_lds[(brow * 10 + c) * 16 + bb], acc[c][j]);  // ds_add_f32
    }
  __syncthreads();
  for (int i = threadIdx.x; i < 2560; i += 256)
    PART[(size_t)blockIdx.x * 2560 + i] = s_lds[i];
}

// ---------------------------------------------------------------------------
// reduce partials across 256 blocks, squash 16-vectors -> v (+ d_out on last)
// ---------------------------------------------------------------------------
__global__ __launch_bounds__(256)
void reduce_squash_kernel(const float* __restrict__ PART, float* __restrict__ V,
                          float* __restrict__ OUT, int write_out) {
  __shared__ float ss[2560];
  for (int q = threadIdx.x; q < 2560; q += 256) {
    float sum = 0.f;
    for (int b = 0; b < 256; ++b) sum += PART[(size_t)b * 2560 + q];
    ss[q] = sum;
  }
  __syncthreads();
  if (threadIdx.x < 160) {
    float sq = 0.f;
#pragma unroll
    for (int o = 0; o < 16; ++o) { const float x = ss[threadIdx.x * 16 + o]; sq += x * x; }
    const float sc = (sq / (1.f + sq)) / sqrtf(sq + 1e-8f);
#pragma unroll
    for (int o = 0; o < 16; ++o) {
      const float vv = ss[threadIdx.x * 16 + o] * sc;
      V[threadIdx.x * 16 + o] = vv;
      if (write_out) OUT[threadIdx.x * 16 + o] = vv;
    }
  }
}

// ---------------------------------------------------------------------------
// b-update: BL[b,c,r] += sum_o u_hat[b,c,r,o] * v[b,c,o]
// recomputes u_hat tiles, reduces over o via shfl_xor across 16 lanes
// ---------------------------------------------------------------------------
__global__ __launch_bounds__(256)
void route_b_kernel(const float* __restrict__ U, const float* __restrict__ RW,
                    const float* __restrict__ V, float* __restrict__ BL) {
  __shared__ float v_lds[2560];
  for (int i = threadIdx.x; i < 2560; i += 256) v_lds[i] = V[i];
  __syncthreads();

  const int lane = threadIdx.x & 31;
  const int wid  = threadIdx.x >> 5;
  const int half = lane >> 4;
  const int bb   = lane & 15;
  const int gw   = blockIdx.x * 8 + wid;

  for (int t = 0; t < 18; ++t) {
    const int r = gw * 18 + t;
    const float* ub = U + (size_t)bb * 294912 + (size_t)r * 8;
    const float2 a1 = *(const float2*)(ub + 2 * half);
    const float2 a2 = *(const float2*)(ub + 4 + 2 * half);
    const v2f A1 = { a1.x, a1.y };
    const v2f A2 = { a2.x, a2.y };
    const float* wb = RW + (size_t)r * 1280;
#pragma unroll
    for (int c = 0; c < 10; ++c) {
      const float* wc = wb + c * 128;
      const int i0 = 2 * half;
      const v2f B1 = { wc[i0 * 16 + bb],       wc[(i0 + 1) * 16 + bb] };
      const v2f B2 = { wc[(i0 + 4) * 16 + bb], wc[(i0 + 5) * 16 + bb] };
      v8f tile = {};
      tile = wmma_f32_k4(A1, B1, tile);
      tile = wmma_f32_k4(A2, B2, tile);
      float p[8];
#pragma unroll
      for (int j = 0; j < 8; ++j) {
        const int brow = j + 8 * half;
        p[j] = tile[j] * v_lds[(brow * 10 + c) * 16 + bb];
      }
#pragma unroll
      for (int m = 1; m < 16; m <<= 1)
#pragma unroll
        for (int j = 0; j < 8; ++j)
          p[j] += __shfl_xor(p[j], m, 32);
      if (bb == 0) {
#pragma unroll
        for (int j = 0; j < 8; ++j) {
          const int brow = j + 8 * half;
          const size_t bi = ((size_t)(brow * 10 + c)) * 36864 + r;
          BL[bi] += p[j];
        }
      }
    }
  }
}

// ---------------------------------------------------------------------------
extern "C" void kernel_launch(void* const* d_in, const int* in_sizes, int n_in,
                              void* d_out, int out_size, void* d_ws, size_t ws_size,
                              hipStream_t stream) {
  (void)in_sizes; (void)n_in; (void)out_size; (void)ws_size;
  const float* X  = (const float*)d_in[0];
  const float* W1 = (const float*)d_in[1];
  const float* B1 = (const float*)d_in[2];
  const float* PW = (const float*)d_in[3];
  const float* PB = (const float*)d_in[4];
  const float* RW = (const float*)d_in[5];
  char* ws = (char*)d_ws;
  float* H  = (float*)(ws);                                          //  6.55 MB
  float* U  = (float*)(ws + 6553600);                                // 18.87 MB
  float* BL = (float*)(ws + 6553600 + 18874368);                     // 23.59 MB
  float* ST = (float*)(ws + 6553600 + 18874368 + 23592960);          //  2 KB
  float* PT = (float*)(ws + 6553600 + 18874368 + 23592960 + 2048);   //  2.62 MB
  float* V  = (float*)(ws + 6553600 + 18874368 + 23592960 + 2048 + 2621440);
  float* OUT = (float*)d_out;

  conv1_kernel<<<6400, 256, 0, stream>>>(X, W1, B1, H);
  prim_gemm_kernel<<<dim3(64, 9), 256, 0, stream>>>(H, PW, PB, U);
  squash_kernel<<<2304, 256, 0, stream>>>(U);
  zero_kernel<<<23040, 256, 0, stream>>>(BL, 5898240);
  for (int it = 0; it < 3; ++it) {
    softmax_stats_kernel<<<160, 256, 0, stream>>>(BL, ST);
    route_s_kernel<<<256, 256, 0, stream>>>(U, RW, BL, ST, PT);
    reduce_squash_kernel<<<1, 256, 0, stream>>>(PT, V, OUT, it == 2 ? 1 : 0);
    if (it < 2) route_b_kernel<<<256, 256, 0, stream>>>(U, RW, V, BL);
  }
}
